// DeepRTE_3410204033244
// MI455X (gfx1250) — matrix operand order repaired and checked by reading, not compiled, not run on hardware
//
#include <hip/hip_runtime.h>
#include <hip/hip_bf16.h>
#include <math.h>

// ---------------- problem constants ----------------
#define B_   2
#define P_   128
#define NB_  64
#define T_   256
#define V_   8
#define HH_  4
#define DK_  16
#define L_   128
#define R2_  (B_*P_*NB_)        // 16384 rows (act0 path)
#define R_   (R2_*V_)           // 131072 rows (s0 path)
#define KF_  96                 // 72 features padded to 3*32

typedef __attribute__((ext_vector_type(16))) __bf16 v16bf;
typedef __attribute__((ext_vector_type(8)))  __bf16 v8bf;
typedef __attribute__((ext_vector_type(8)))  float  v8f;

// =====================================================================
// Weight convert: f32 W[K x N] (row major) -> bf16 Wt[N x Kp], zero pad
// =====================================================================
__global__ void cvtw_kernel(const float* __restrict__ W, __bf16* __restrict__ Wt,
                            int K, int N, int Kp)
{
    int o = blockIdx.x * blockDim.x + threadIdx.x;
    if (o >= N * Kp) return;
    int k = o % Kp, n = o / Kp;
    Wt[o] = (k < K) ? (__bf16)W[k * N + n] : (__bf16)0.0f;
}

// =====================================================================
// v_heads[b,t,0:64] = sigma[b,t,:2] @ wv + bv
// =====================================================================
__global__ void vheads_kernel(const float* __restrict__ sigma,
                              const float* __restrict__ wv, const float* __restrict__ bv,
                              float* __restrict__ vh)
{
    int o = blockIdx.x * blockDim.x + threadIdx.x;
    if (o >= B_ * T_ * 64) return;
    int j = o & 63; int bt = o >> 6;
    vh[o] = sigma[bt * 2 + 0] * wv[j] + sigma[bt * 2 + 1] * wv[64 + j] + bv[j];
}

// =====================================================================
// Attention: one block per (b,p).  36 head instances:
//   0..3   : main heads (q from phase_coords)        -> att   [b,p,64]
//   4+4v+h : velocity-resolved (q from cs[v])        -> att_s [b,p,v,64]
// =====================================================================
__global__ __launch_bounds__(256) void attention_kernel(
    const float* __restrict__ phase_coords, const float* __restrict__ position_coords,
    const float* __restrict__ velocity_coords,
    const float* __restrict__ wq, const float* __restrict__ bq,
    const float* __restrict__ wk, const float* __restrict__ bk,
    const float* __restrict__ wo, const float* __restrict__ bo,
    const float* __restrict__ vh,
    float* __restrict__ att, float* __restrict__ atts)
{
    __shared__ float rel[T_][2];
    __shared__ float qall[36][16];
    __shared__ float attn36[36][16];
    __shared__ float pc4[4];

    const int bp = blockIdx.x;
    const int b  = bp / P_;
    const int tid = threadIdx.x;

    if (tid < 4) pc4[tid] = phase_coords[bp * 4 + tid];
    __syncthreads();
    const float px = pc4[0], py = pc4[1], vx = pc4[2], vy = pc4[3];

    { // relative coordinates for every t
        int t = tid;
        rel[t][0] = position_coords[(b * T_ + t) * 2 + 0] - px;
        rel[t][1] = position_coords[(b * T_ + t) * 2 + 1] - py;
    }
    // query vectors for all 36 instances
    for (int idx = tid; idx < 36 * 16; idx += 256) {
        int inst = idx >> 4, d = idx & 15;
        float c2, c3; int h;
        if (inst < 4) { h = inst; c2 = vx; c3 = vy; }
        else { int v = (inst - 4) >> 2; h = (inst - 4) & 3;
               c2 = velocity_coords[(b * V_ + v) * 2 + 0];
               c3 = velocity_coords[(b * V_ + v) * 2 + 1]; }
        int col = h * 16 + d;
        qall[inst][d] = px * wq[0 * 64 + col] + py * wq[1 * 64 + col]
                      + c2 * wq[2 * 64 + col] + c3 * wq[3 * 64 + col] + bq[col];
    }
    __syncthreads();

    const int wave = tid >> 5, lane = tid & 31;
    const float scale = 0.25f; // 1/sqrt(DK)

    for (int inst = wave; inst < 36; inst += 8) {
        int h; float mvx, mvy;
        if (inst < 4) { h = inst; mvx = vx; mvy = vy; }
        else { int v = (inst - 4) >> 2; h = (inst - 4) & 3;
               mvx = velocity_coords[(b * V_ + v) * 2 + 0];
               mvy = velocity_coords[(b * V_ + v) * 2 + 1]; }

        float logit[8], mx = -3.0e38f;
        #pragma unroll
        for (int j = 0; j < 8; j++) {
            int t = lane + j * 32;
            float rx = rel[t][0], ry = rel[t][1];
            float ptx = rx + px, pty = ry + py;   // position_coords[t]
            float l = 0.f;
            #pragma unroll
            for (int d = 0; d < 16; d++) {
                int col = h * 16 + d;
                float kd = ptx * wk[0 * 64 + col] + pty * wk[1 * 64 + col]
                         + rx  * wk[2 * 64 + col] + ry  * wk[3 * 64 + col] + bk[col];
                l += qall[inst][d] * kd;
            }
            l *= scale;
            bool ok = (rx * mvx + ry * mvy) <= 0.f;
            logit[j] = ok ? l : -1.0e30f;
            mx = fmaxf(mx, logit[j]);
        }
        for (int o = 16; o; o >>= 1) mx = fmaxf(mx, __shfl_xor(mx, o));
        float sume = 0.f, w8[8];
        #pragma unroll
        for (int j = 0; j < 8; j++) { w8[j] = __expf(logit[j] - mx); sume += w8[j]; }
        for (int o = 16; o; o >>= 1) sume += __shfl_xor(sume, o);

        float accd[16];
        #pragma unroll
        for (int d = 0; d < 16; d++) accd[d] = 0.f;
        #pragma unroll
        for (int j = 0; j < 8; j++) {
            int t = lane + j * 32;
            const float* vrow = &vh[(size_t)(b * T_ + t) * 64 + h * 16];
            #pragma unroll
            for (int d = 0; d < 16; d++) accd[d] += w8[j] * vrow[d];
        }
        #pragma unroll
        for (int d = 0; d < 16; d++)
            for (int o = 16; o; o >>= 1) accd[d] += __shfl_xor(accd[d], o);
        if (lane == 0) {
            float inv = 1.f / sume;
            #pragma unroll
            for (int d = 0; d < 16; d++) attn36[inst][d] = accd[d] * inv;
        }
    }
    __syncthreads();

    // project through wo, exp(-x)
    for (int idx = tid; idx < 9 * 64; idx += 256) {
        int grp = idx >> 6, j = idx & 63;
        const float (*src)[16] = (grp == 0) ? &attn36[0] : &attn36[4 + (grp - 1) * 4];
        float s = bo[j];
        #pragma unroll
        for (int r = 0; r < 64; r++) s += src[r >> 4][r & 15] * wo[r * 64 + j];
        float e = __expf(-s);
        if (grp == 0) att[(size_t)bp * 64 + j] = e;
        else          atts[((size_t)bp * V_ + (grp - 1)) * 64 + j] = e;
    }
}

// =====================================================================
// Feature builders (bf16, K padded to 96)
// =====================================================================
__global__ void buildXa_kernel(const float* __restrict__ phase_coords,
                               const float* __restrict__ boundary_coords,
                               const float* __restrict__ att, __bf16* __restrict__ X)
{
    long o = (long)blockIdx.x * blockDim.x + threadIdx.x;
    if (o >= (long)R2_ * KF_) return;
    int c = (int)(o % KF_); long r = o / KF_;
    int n = (int)(r % NB_); long bpp = r / NB_;      // b*P+p
    int b = (int)(bpp / P_);
    float v;
    if      (c < 4)  v = phase_coords[bpp * 4 + c];
    else if (c < 8)  v = boundary_coords[(b * NB_ + n) * 4 + (c - 4)];
    else if (c < 72) v = att[bpp * 64 + (c - 8)];
    else             v = 0.f;
    X[o] = (__bf16)v;
}

__global__ void buildXs_kernel(const float* __restrict__ phase_coords,
                               const float* __restrict__ velocity_coords,
                               const float* __restrict__ boundary_coords,
                               const float* __restrict__ atts, __bf16* __restrict__ X)
{
    long o = (long)blockIdx.x * blockDim.x + threadIdx.x;
    if (o >= (long)R_ * KF_) return;
    int c = (int)(o % KF_); long r = o / KF_;
    int v = (int)(r % V_);  long rn = r / V_;
    int n = (int)(rn % NB_); long bpp = rn / NB_;
    int b = (int)(bpp / P_);
    float x;
    if      (c < 2)  x = phase_coords[bpp * 4 + c];                          // pos
    else if (c < 4)  x = velocity_coords[(b * V_ + v) * 2 + (c - 2)];
    else if (c < 8)  x = boundary_coords[(b * NB_ + n) * 4 + (c - 4)];
    else if (c < 72) x = atts[(bpp * V_ + v) * 64 + (c - 8)];
    else             x = 0.f;
    X[o] = (__bf16)x;
}

// =====================================================================
// WMMA bf16 GEMM:  C[M x 128] = tanh(A[M x KP] @ Wt^T + bias)
// Wt is [128 x KP] bf16 (pre-transposed), staged in LDS.
// Block = 8 waves, 256x128 output tile; wave w owns rows {w*16..w*16+15}
// and {128+w*16..}.  Per k-step: 16 ds_load_b128 -> one wait -> 16 WMMAs
// (each B fragment reused by two M tiles).
// =====================================================================
template <int KP>
__global__ __launch_bounds__(256) void gemm_bias_tanh(
    const __bf16* __restrict__ A, const __bf16* __restrict__ Wt,
    const float* __restrict__ bias, __bf16* __restrict__ C)
{
    __shared__ __bf16 lWt[128 * KP];
    const int tid = threadIdx.x;
    for (int i = tid * 8; i < 128 * KP; i += 256 * 8)
        *(v8bf*)&lWt[i] = *(const v8bf*)&Wt[i];
    __syncthreads();

    const int wave = tid >> 5, lane = tid & 31;
    const int lo16 = lane & 15, hi = lane >> 4;
    const long mbase0 = (long)blockIdx.x * 256 + wave * 16;
    const long mbase1 = mbase0 + 128;
    const __bf16* arow0 = A + (mbase0 + lo16) * KP;
    const __bf16* arow1 = A + (mbase1 + lo16) * KP;

    v8f acc0[8], acc1[8];
    #pragma unroll
    for (int i = 0; i < 8; i++) {
        acc0[i] = (v8f){0, 0, 0, 0, 0, 0, 0, 0};
        acc1[i] = (v8f){0, 0, 0, 0, 0, 0, 0, 0};
    }

    #pragma unroll
    for (int kb = 0; kb < KP; kb += 32) {
        // A fragments: row lo16, K chunks {hi*8..hi*8+7} and {+16}
        union { v16bf v; v8bf h[2]; } af0, af1;
        af0.h[0] = *(const v8bf*)&arow0[kb + hi * 8];
        af0.h[1] = *(const v8bf*)&arow0[kb + hi * 8 + 16];
        af1.h[0] = *(const v8bf*)&arow1[kb + hi * 8];
        af1.h[1] = *(const v8bf*)&arow1[kb + hi * 8 + 16];
        // preload ALL 8 B fragments for this k-step (batched ds_loads)
        union { v16bf v; v8bf h[2]; } bfg[8];
        #pragma unroll
        for (int nt = 0; nt < 8; nt++) {
            const __bf16* wrow = &lWt[(nt * 16 + lo16) * KP + kb + hi * 8];
            bfg[nt].h[0] = *(const v8bf*)&wrow[0];
            bfg[nt].h[1] = *(const v8bf*)&wrow[16];
        }
        // back-to-back WMMA burst; each B fragment used twice
        #pragma unroll
        for (int nt = 0; nt < 8; nt++)
            acc0[nt] = __builtin_amdgcn_wmma_f32_16x16x32_bf16(
                false, af0.v, false, bfg[nt].v, (short)0, acc0[nt], false, false);
        #pragma unroll
        for (int nt = 0; nt < 8; nt++)
            acc1[nt] = __builtin_amdgcn_wmma_f32_16x16x32_bf16(
                false, af1.v, false, bfg[nt].v, (short)0, acc1[nt], false, false);
    }

    #pragma unroll
    for (int nt = 0; nt < 8; nt++) {
        const int col = nt * 16 + lo16;
        const float bcol = bias[col];
        #pragma unroll
        for (int i = 0; i < 8; i++) {
            long row0 = mbase0 + hi * 8 + i;
            long row1 = mbase1 + hi * 8 + i;
            C[row0 * 128 + col] = (__bf16)tanhf(acc0[nt][i] + bcol);
            C[row1 * 128 + col] = (__bf16)tanhf(acc1[nt][i] + bcol);
        }
    }
}

// =====================================================================
// y[b,p,n,w,d] = sum_v (-ssk[b,w,v]*vw[b,v]) * cur[b,p,n,v,d]
// =====================================================================
__global__ void mix_kernel(const __bf16* __restrict__ cur, const float* __restrict__ ssk,
                           const float* __restrict__ vw, __bf16* __restrict__ outm)
{
    long o = (long)blockIdx.x * blockDim.x + threadIdx.x;
    if (o >= (long)R_ * 128) return;
    int d = (int)(o & 127); long r = o >> 7;
    int w = (int)(r & (V_ - 1)); long base = r - w;
    int b = (int)(r / ((long)P_ * NB_ * V_));
    float acc = 0.f;
    #pragma unroll
    for (int v = 0; v < V_; v++) {
        float k = -ssk[(b * V_ + w) * V_ + v] * vw[b * V_ + v];
        acc += k * (float)cur[(base + v) * 128 + d];
    }
    outm[o] = (__bf16)acc;
}

// y2[b,p,n,d] = sum_v (-sk[b,p,v]*vw[b,v]) * cur[b,p,n,v,d]
__global__ void contract_kernel(const __bf16* __restrict__ cur, const float* __restrict__ sk,
                                const float* __restrict__ vw, __bf16* __restrict__ outc)
{
    long o = (long)blockIdx.x * blockDim.x + threadIdx.x;
    if (o >= (long)R2_ * 128) return;
    int d = (int)(o & 127); long r2 = o >> 7;     // (b*P+p)*NB+n
    long bpp = r2 / NB_;
    int b = (int)(bpp / P_);
    float acc = 0.f;
    #pragma unroll
    for (int v = 0; v < V_; v++) {
        float k = -sk[bpp * V_ + v] * vw[b * V_ + v];
        acc += k * (float)cur[(r2 * V_ + v) * 128 + d];
    }
    outc[o] = (__bf16)acc;
}

// =====================================================================
// cur = LayerNorm(s0 + y) * scale + offset   (one 128-thread block / row)
// =====================================================================
__global__ __launch_bounds__(128) void resln_kernel(
    const __bf16* __restrict__ s0, const __bf16* __restrict__ y,
    const float* __restrict__ scale, const float* __restrict__ offset,
    __bf16* __restrict__ outc)
{
    const long row = blockIdx.x;
    const int tid = threadIdx.x;
    __shared__ float rsum[4], rss[4];
    float x = (float)s0[row * 128 + tid] + (float)y[row * 128 + tid];
    float s = x, ss = x * x;
    for (int o = 16; o; o >>= 1) { s += __shfl_xor(s, o); ss += __shfl_xor(ss, o); }
    int wave = tid >> 5;
    if ((tid & 31) == 0) { rsum[wave] = s; rss[wave] = ss; }
    __syncthreads();
    float tot = rsum[0] + rsum[1] + rsum[2] + rsum[3];
    float tss = rss[0] + rss[1] + rss[2] + rss[3];
    float m = tot * (1.f / 128.f);
    float var = tss * (1.f / 128.f) - m * m;
    float rs = rsqrtf(var + 1e-5f);
    outc[row * 128 + tid] = (__bf16)((x - m) * rs * scale[tid] + offset[tid]);
}

// =====================================================================
// Final: act_out = LN(act0+y); green = exp(act_out@proj_w);
//        out[b,p] = sum_n green * boundary * boundary_weights
// =====================================================================
__global__ __launch_bounds__(256) void final_kernel(
    const __bf16* __restrict__ act0, const __bf16* __restrict__ y,
    const float* __restrict__ ln_scale, const float* __restrict__ ln_offset,
    const float* __restrict__ proj_w, const float* __restrict__ boundary,
    const float* __restrict__ bw, float* __restrict__ out)
{
    const int bp = blockIdx.x, b = bp / P_;
    const int tid = threadIdx.x, wave = tid >> 5, lane = tid & 31;
    __shared__ float partial[8];
    float accw = 0.f;
    for (int n = wave; n < NB_; n += 8) {
        size_t row = (size_t)bp * NB_ + n;
        float x[4], s = 0.f, ss = 0.f;
        #pragma unroll
        for (int j = 0; j < 4; j++) {
            int c = lane * 4 + j;
            x[j] = (float)act0[row * 128 + c] + (float)y[row * 128 + c];
            s += x[j]; ss += x[j] * x[j];
        }
        for (int o = 16; o; o >>= 1) { s += __shfl_xor(s, o); ss += __shfl_xor(ss, o); }
        float m = s * (1.f / 128.f);
        float var = ss * (1.f / 128.f) - m * m;
        float rs = rsqrtf(var + 1e-5f);
        float dot = 0.f;
        #pragma unroll
        for (int j = 0; j < 4; j++) {
            int c = lane * 4 + j;
            float a = (x[j] - m) * rs * ln_scale[c] + ln_offset[c];
            dot += a * proj_w[c];
        }
        for (int o = 16; o; o >>= 1) dot += __shfl_xor(dot, o);
        accw += __expf(dot) * boundary[b * NB_ + n] * bw[b * NB_ + n];
    }
    if (lane == 0) partial[wave] = accw;
    __syncthreads();
    if (tid == 0) {
        float tot = 0.f;
        for (int w = 0; w < 8; w++) tot += partial[w];
        out[bp] = tot;
    }
}

// =====================================================================
// host-side launch
// =====================================================================
extern "C" void kernel_launch(void* const* d_in, const int* in_sizes, int n_in,
                              void* d_out, int out_size, void* d_ws, size_t ws_size,
                              hipStream_t stream) {
    const float* phase_coords    = (const float*)d_in[0];
    const float* boundary_coords = (const float*)d_in[1];
    const float* boundary        = (const float*)d_in[2];
    const float* boundary_w      = (const float*)d_in[3];
    const float* position_coords = (const float*)d_in[4];
    const float* sigma           = (const float*)d_in[5];
    const float* velocity_coords = (const float*)d_in[6];
    const float* velocity_w      = (const float*)d_in[7];
    const float* scattering_k    = (const float*)d_in[8];
    const float* self_scat_k     = (const float*)d_in[9];
    const float* wq = (const float*)d_in[10]; const float* bq = (const float*)d_in[11];
    const float* wk = (const float*)d_in[12]; const float* bk = (const float*)d_in[13];
    const float* wv = (const float*)d_in[14]; const float* bv = (const float*)d_in[15];
    const float* wo = (const float*)d_in[16]; const float* bo = (const float*)d_in[17];
    const float* att_l1_w  = (const float*)d_in[18]; const float* att_l1_b  = (const float*)d_in[19];
    const float* att_l2_w  = (const float*)d_in[20]; const float* att_l2_b  = (const float*)d_in[21];
    const float* att_out_w = (const float*)d_in[22]; const float* att_out_b = (const float*)d_in[23];
    const float* sc_w = (const float*)d_in[24]; const float* sc_b = (const float*)d_in[25];
    const float* sc_ln_scale = (const float*)d_in[26]; const float* sc_ln_offset = (const float*)d_in[27];
    const float* sc_out_w = (const float*)d_in[28]; const float* sc_out_b = (const float*)d_in[29];
    const float* out_ln_scale = (const float*)d_in[30]; const float* out_ln_offset = (const float*)d_in[31];
    const float* proj_w = (const float*)d_in[32];
    float* out = (float*)d_out;

    // ---- workspace carve-up (256B aligned) ----
    char* ws = (char*)d_ws;
    size_t off = 0;
    auto alloc = [&](size_t bytes) { size_t o = off; off = (off + bytes + 255) & ~(size_t)255; return o; };
    float*  vh    = (float*) (ws + alloc((size_t)B_*T_*64*4));
    float*  attB  = (float*) (ws + alloc((size_t)B_*P_*64*4));
    float*  attsB = (float*) (ws + alloc((size_t)B_*P_*V_*64*4));
    __bf16* Wt1   = (__bf16*)(ws + alloc((size_t)128*KF_*2));
    __bf16* Wt2   = (__bf16*)(ws + alloc((size_t)128*128*2));
    __bf16* Wt3   = (__bf16*)(ws + alloc((size_t)128*128*2));
    __bf16* Wsc0  = (__bf16*)(ws + alloc((size_t)128*128*2));
    __bf16* Wsc1  = (__bf16*)(ws + alloc((size_t)128*128*2));
    __bf16* Wso   = (__bf16*)(ws + alloc((size_t)128*128*2));
    __bf16* Xa    = (__bf16*)(ws + alloc((size_t)R2_*KF_*2));
    __bf16* Xs    = (__bf16*)(ws + alloc((size_t)R_*KF_*2));
    __bf16* ah1   = (__bf16*)(ws + alloc((size_t)R2_*128*2));
    __bf16* ah2   = (__bf16*)(ws + alloc((size_t)R2_*128*2));
    __bf16* act0b = (__bf16*)(ws + alloc((size_t)R2_*128*2));
    __bf16* yh1   = (__bf16*)(ws + alloc((size_t)R2_*128*2));
    __bf16* yh2   = (__bf16*)(ws + alloc((size_t)R2_*128*2));
    __bf16* bufA  = (__bf16*)(ws + alloc((size_t)R_*128*2));
    __bf16* bufB  = (__bf16*)(ws + alloc((size_t)R_*128*2));
    __bf16* s0    = (__bf16*)(ws + alloc((size_t)R_*128*2));
    __bf16* cur   = (__bf16*)(ws + alloc((size_t)R_*128*2));
    (void)ws_size; (void)in_sizes; (void)n_in; (void)out_size;

    // ---- weight conversion (transposed bf16) ----
    auto cvt = [&](const float* W, __bf16* dst, int K, int N, int Kp) {
        int n = N * Kp;
        cvtw_kernel<<<(n + 255) / 256, 256, 0, stream>>>(W, dst, K, N, Kp);
    };
    cvt(att_l1_w,            Wt1, 72, 128, KF_);
    cvt(att_l2_w,            Wt2, 128, 128, 128);
    cvt(att_out_w,           Wt3, 128, 128, 128);
    cvt(sc_w,                Wsc0, 128, 128, 128);
    cvt(sc_w + 128 * 128,    Wsc1, 128, 128, 128);
    cvt(sc_out_w,            Wso, 128, 128, 128);

    // ---- attention pipeline ----
    vheads_kernel<<<(B_*T_*64)/256, 256, 0, stream>>>(sigma, wv, bv, vh);
    attention_kernel<<<B_*P_, 256, 0, stream>>>(
        phase_coords, position_coords, velocity_coords,
        wq, bq, wk, bk, wo, bo, vh, attB, attsB);

    // ---- feature matrices ----
    buildXa_kernel<<<((long)R2_*KF_ + 255)/256, 256, 0, stream>>>(
        phase_coords, boundary_coords, attB, Xa);
    buildXs_kernel<<<((long)R_*KF_ + 255)/256, 256, 0, stream>>>(
        phase_coords, velocity_coords, boundary_coords, attsB, Xs);

    // ---- act0 path MLP (R2 rows, 256 rows/block) ----
    gemm_bias_tanh<KF_><<<R2_/256, 256, 0, stream>>>(Xa,  Wt1, att_l1_b,  ah1);
    gemm_bias_tanh<128><<<R2_/256, 256, 0, stream>>>(ah1, Wt2, att_l2_b,  ah2);
    gemm_bias_tanh<128><<<R2_/256, 256, 0, stream>>>(ah2, Wt3, att_out_b, act0b);

    // ---- s0 path MLP (R rows) ----
    gemm_bias_tanh<KF_><<<R_/256, 256, 0, stream>>>(Xs,   Wt1, att_l1_b,  bufA);
    gemm_bias_tanh<128><<<R_/256, 256, 0, stream>>>(bufA, Wt2, att_l2_b,  bufB);
    gemm_bias_tanh<128><<<R_/256, 256, 0, stream>>>(bufB, Wt3, att_out_b, s0);

    // ---- scattering loop (2 iterations) ----
    const long mixN = (long)R_ * 128;
    // i = 0 (cur == s0)
    mix_kernel<<<mixN/256, 256, 0, stream>>>(s0, self_scat_k, velocity_w, bufA);
    gemm_bias_tanh<128><<<R_/256, 256, 0, stream>>>(bufA, Wsc0, sc_b, bufB);
    resln_kernel<<<R_, 128, 0, stream>>>(s0, bufB, sc_ln_scale, sc_ln_offset, cur);
    // i = 1
    mix_kernel<<<mixN/256, 256, 0, stream>>>(cur, self_scat_k, velocity_w, bufA);
    gemm_bias_tanh<128><<<R_/256, 256, 0, stream>>>(bufA, Wsc1, sc_b + 128, bufB);
    resln_kernel<<<R_, 128, 0, stream>>>(s0, bufB, sc_ln_scale + 128, sc_ln_offset + 128, cur);

    // ---- output contraction + final GEMM + green reduction ----
    contract_kernel<<<((long)R2_*128)/256, 256, 0, stream>>>(cur, scattering_k, velocity_w, yh1);
    gemm_bias_tanh<128><<<R2_/256, 256, 0, stream>>>(yh1, Wso, sc_out_b, yh2);
    final_kernel<<<B_*P_, 256, 0, stream>>>(act0b, yh2, out_ln_scale, out_ln_offset,
                                            proj_w, boundary, boundary_w, out);
}
